// MQOTLayer_84559316124306
// MI455X (gfx1250) — matrix operand order repaired and checked by reading, not compiled, not run on hardware
//
#include <hip/hip_runtime.h>
#include <hip/hip_bf16.h>

typedef __attribute__((ext_vector_type(16))) _Float16 v16h;
typedef __attribute__((ext_vector_type(8)))  _Float16 v8h;
typedef __attribute__((ext_vector_type(8)))  float    v8f;

constexpr int   B_    = 16;
constexpr int   Ta_   = 2048;
constexpr int   Tv_   = 1024;
constexpr int   D_    = 512;
constexpr float EPS_  = 0.15f;
constexpr int   NIT_  = 20;

__device__ __forceinline__ float waveReduceSum(float s) {
#pragma unroll
  for (int off = 16; off > 0; off >>= 1) s += __shfl_xor(s, off, 32);
  return s;
}

// ---------------- Phase 1: L2-normalize rows, emit fp16 ----------------
// One wave32 per row of D=512: 16 floats per lane, butterfly reduce.
__global__ void __launch_bounds__(256)
normalize_rows(const float* __restrict__ src, _Float16* __restrict__ dst,
               int nrows) {
  const int lane = threadIdx.x & 31;
  const int wave = threadIdx.x >> 5;
  const int row  = blockIdx.x * 8 + wave;
  if (row >= nrows) return;
  const float* p = src + (size_t)row * D_;
  float x[16];
  float ss = 0.f;
#pragma unroll
  for (int t = 0; t < 16; ++t) { x[t] = p[t * 32 + lane]; ss += x[t] * x[t]; }
  ss = waveReduceSum(ss);
  const float scale = 1.0f / fmaxf(sqrtf(ss), 1e-12f);
  _Float16* q = dst + (size_t)row * D_;
#pragma unroll
  for (int t = 0; t < 16; ++t) q[t * 32 + lane] = (_Float16)(x[t] * scale);
}

// ---------------- Phase 2: WMMA GEMM + fused cost -> K = -cost/eps -----
// 2x2 register-blocked: each wave owns a 32x32 tile of sim[b] (4 f32
// accumulators), so each k-step loads 2 A-frags + 2 B-frags and issues
// 4 WMMAs -> 2x the operand reuse of a single-tile wave.
// A fragment (16x32 f16): lane m=(l&15), halves [k0+8*(l>>4) .. +7] and +16.
// B fragment (32x16 f16): lane n=(l&15), halves [k0+8*(l>>4) .. +7] and +16
//   of the (normalized) video row n.
__device__ __forceinline__ v16h load_afrag(const _Float16* p) {
  const v8h a0 = *(const v8h*)(p);
  const v8h a1 = *(const v8h*)(p + 16);
  v16h af;
#pragma unroll
  for (int i = 0; i < 8; ++i) { af[i] = a0[i]; af[i + 8] = a1[i]; }
  return af;
}

__global__ void __launch_bounds__(256)
cost_gemm(const _Float16* __restrict__ aH, const _Float16* __restrict__ vH,
          const float* __restrict__ quality, const float* __restrict__ lt_p,
          const float* __restrict__ lq_p, float* __restrict__ Kout) {
  const int lane  = threadIdx.x & 31;
  const int wave  = threadIdx.x >> 5;
  const int b     = blockIdx.z;
  const int tileM = blockIdx.y * 64  + (wave >> 2) * 32;  // 2 waves in M
  const int tileN = blockIdx.x * 128 + (wave & 3) * 32;   // 4 waves in N

  const float lt = lt_p[0];
  const float lq = lq_p[0];

  const int mRow = tileM + (lane & 15);
  const int nCol = tileN + (lane & 15);
  const _Float16* aBase0 = aH + (size_t)(b * Ta_ + mRow) * D_ + ((lane >> 4) << 3);
  const _Float16* aBase1 = aBase0 + (size_t)16 * D_;
  const _Float16* bBase0 = vH + (size_t)(b * Tv_ + nCol) * D_ + ((lane >> 4) << 3);
  const _Float16* bBase1 = bBase0 + (size_t)16 * D_;

  v8f c00 = {}, c01 = {}, c10 = {}, c11 = {};
#pragma unroll 2
  for (int k0 = 0; k0 < D_; k0 += 32) {
    const v16h af0 = load_afrag(aBase0 + k0);
    const v16h af1 = load_afrag(aBase1 + k0);
    const v16h bf0 = load_afrag(bBase0 + k0);
    const v16h bf1 = load_afrag(bBase1 + k0);
    c00 = __builtin_amdgcn_wmma_f32_16x16x32_f16(false, af0, false, bf0,
                                                 (short)0, c00, false, false);
    c01 = __builtin_amdgcn_wmma_f32_16x16x32_f16(false, af0, false, bf1,
                                                 (short)0, c01, false, false);
    c10 = __builtin_amdgcn_wmma_f32_16x16x32_f16(false, af1, false, bf0,
                                                 (short)0, c10, false, false);
    c11 = __builtin_amdgcn_wmma_f32_16x16x32_f16(false, af1, false, bf1,
                                                 (short)0, c11, false, false);
  }

  // Epilogue: C/D layout -> VGPR r holds (M = r + 8*(lane>>4), N = lane&15).
  const int jLane = lane & 15;
  const int iBase = (lane >> 4) << 3;
#pragma unroll
  for (int nt = 0; nt < 2; ++nt) {
    const int   j     = tileN + nt * 16 + jLane;
    const float qterm = lq * (1.0f - quality[b * Tv_ + j]);
    const float jf    = (float)j * (1.0f / (float)Tv_);
#pragma unroll
    for (int mt = 0; mt < 2; ++mt) {
      const v8f& c  = (mt == 0) ? (nt == 0 ? c00 : c01) : (nt == 0 ? c10 : c11);
      const int  i0 = tileM + mt * 16 + iBase;
#pragma unroll
      for (int r = 0; r < 8; ++r) {
        const int   i    = i0 + r;
        const float td   = fabsf((float)i * (1.0f / (float)Ta_) - jf);
        const float cost = (1.0f - c[r]) + lt * td + qterm;
        Kout[(size_t)(b * Ta_ + i) * Tv_ + j] = cost * (-1.0f / EPS_);
      }
    }
  }
}

// ---------------- Phase 3a: u = -logsumexp_j(K + v) --------------------
// One wave32 per row (Tv=1024 -> 32 elems/lane), online logsumexp.
__global__ void __launch_bounds__(256)
row_lse(const float* __restrict__ Kmat, const float* __restrict__ vv,
        float* __restrict__ u) {
  const int lane = threadIdx.x & 31;
  const int wave = threadIdx.x >> 5;
  const int row  = blockIdx.x * 8 + wave;   // [0, B*Ta)
  const int b    = row >> 11;               // / Ta_
  const float* kp = Kmat + (size_t)row * Tv_;
  const float* vp = vv + b * Tv_;
  float m = -INFINITY, s = 0.f;
#pragma unroll 8
  for (int t = 0; t < 32; ++t) {
    const float x = kp[t * 32 + lane] + vp[t * 32 + lane];
    if (x > m) { s = s * __expf(m - x) + 1.0f; m = x; }
    else       { s += __expf(x - m); }
  }
#pragma unroll
  for (int off = 16; off > 0; off >>= 1) {
    const float om = __shfl_xor(m, off, 32);
    const float os = __shfl_xor(s, off, 32);
    const float M  = fmaxf(m, om);
    s = s * __expf(m - M) + os * __expf(om - M);
    m = M;
  }
  if (lane == 0) u[row] = -(m + __logf(s));
}

// ---------------- Phase 3b: v = -logsumexp_i(K + u) --------------------
// 64 columns x 4 row-chunks per 256-thread block; coalesced over j.
__global__ void __launch_bounds__(256)
col_lse(const float* __restrict__ Kmat, const float* __restrict__ u,
        float* __restrict__ vv) {
  __shared__ float sm[256];
  __shared__ float ss[256];
  const int tx = threadIdx.x & 63;
  const int ty = threadIdx.x >> 6;
  const int b  = blockIdx.y;
  const int j  = blockIdx.x * 64 + tx;
  const float* kp = Kmat + (size_t)(b * Ta_) * Tv_ + j;
  const float* up = u + b * Ta_;
  float m = -INFINITY, s = 0.f;
  const int i0 = ty * (Ta_ / 4);
  for (int t = 0; t < Ta_ / 4; ++t) {
    const int   i = i0 + t;
    const float x = kp[(size_t)i * Tv_] + up[i];
    if (x > m) { s = s * __expf(m - x) + 1.0f; m = x; }
    else       { s += __expf(x - m); }
  }
  sm[threadIdx.x] = m;
  ss[threadIdx.x] = s;
  __syncthreads();
  if (ty == 0) {
#pragma unroll
    for (int p = 1; p < 4; ++p) {
      const float om = sm[tx + p * 64];
      const float os = ss[tx + p * 64];
      const float M  = fmaxf(m, om);
      s = s * __expf(m - M) + os * __expf(om - M);
      m = M;
    }
    vv[b * Tv_ + j] = -(m + __logf(s));
  }
}

// ---------------- Phase 4: transport = exp(K+u+v), row-normalize, in place
__global__ void __launch_bounds__(256)
transport_finalize(float* __restrict__ Kmat, const float* __restrict__ u,
                   const float* __restrict__ vv) {
  const int lane = threadIdx.x & 31;
  const int wave = threadIdx.x >> 5;
  const int row  = blockIdx.x * 8 + wave;
  const int b    = row >> 11;
  float* kp = Kmat + (size_t)row * Tv_;
  const float* vp = vv + b * Tv_;
  const float  ui = u[row];
  float e[32];
  float s = 0.f;
#pragma unroll
  for (int t = 0; t < 32; ++t) {
    e[t] = __expf(kp[t * 32 + lane] + ui + vp[t * 32 + lane]);
    s += e[t];
  }
  s = waveReduceSum(s);
  const float inv = 1.0f / (s + 1e-8f);
#pragma unroll
  for (int t = 0; t < 32; ++t) kp[t * 32 + lane] = e[t] * inv;
}

extern "C" void kernel_launch(void* const* d_in, const int* in_sizes, int n_in,
                              void* d_out, int out_size, void* d_ws, size_t ws_size,
                              hipStream_t stream) {
  const float* audio   = (const float*)d_in[0];
  const float* video   = (const float*)d_in[1];
  const float* quality = (const float*)d_in[2];
  const float* lt      = (const float*)d_in[3];
  const float* lq      = (const float*)d_in[4];
  float* Kmat = (float*)d_out;   // K built here, finalized to transport in place

  char* ws = (char*)d_ws;
  _Float16* aH = (_Float16*)ws;                                   // 32 MB
  _Float16* vH = (_Float16*)(ws + (size_t)B_ * Ta_ * D_ * 2);     // 16 MB
  float*    u  = (float*)(ws + (size_t)B_ * Ta_ * D_ * 2
                             + (size_t)B_ * Tv_ * D_ * 2);        // 128 KB
  float*    vv = u + B_ * Ta_;                                    // 64 KB

  hipMemsetAsync(vv, 0, (size_t)B_ * Tv_ * sizeof(float), stream);

  normalize_rows<<<B_ * Ta_ / 8, 256, 0, stream>>>(audio, aH, B_ * Ta_);
  normalize_rows<<<B_ * Tv_ / 8, 256, 0, stream>>>(video, vH, B_ * Tv_);

  dim3 gGemm(Tv_ / 128, Ta_ / 64, B_);
  cost_gemm<<<gGemm, 256, 0, stream>>>(aH, vH, quality, lt, lq, Kmat);

  for (int it = 0; it < NIT_; ++it) {
    row_lse<<<B_ * Ta_ / 8, 256, 0, stream>>>(Kmat, vv, u);
    col_lse<<<dim3(Tv_ / 64, B_), 256, 0, stream>>>(Kmat, u, vv);
  }

  transport_finalize<<<B_ * Ta_ / 8, 256, 0, stream>>>(Kmat, u, vv);
}
